// Seq2SeqAttn_13391708029128
// MI455X (gfx1250) — matrix operand order repaired and checked by reading, not compiled, not run on hardware
//
#include <hip/hip_runtime.h>
#include <hip/hip_bf16.h>
#include <math.h>

typedef __attribute__((ext_vector_type(16))) _Float16 v16h;
typedef __attribute__((ext_vector_type(8)))  _Float16 v8h;
typedef __attribute__((ext_vector_type(8)))  float    v8f;

#define B_SZ   32
#define T_ENCL 64
#define T_DECL 32
#define EMB    512
#define ENCH   512
#define DECH   1024
#define VOUT   32000

__device__ __forceinline__ float sigm(float x) { return 1.0f / (1.0f + __expf(-x)); }

// ---------------------------------------------------------------------------
// WMMA GEMM, all-f16 operands, f32 accumulate:
//   D[M,N] = act( A[M,K](f16) * W[N,K]^T(f16) + Cin + bias )
// One wave computes a 16x64 strip: 4 accumulators, A-fragment reused by 4
// v_wmma_f32_16x16x32_f16 per K-step. Fragment loads are contiguous 16B/32B
// chunks per lane -> global_load_b128, no cvt in the hot loop.
// permRows: row r=(t*32+b) scattered to out[(b*T_DEC + t)*N + n] (classifier).
// ---------------------------------------------------------------------------
__global__ void wmma_gemm_f16(const _Float16* __restrict__ A,
                              const _Float16* __restrict__ W,
                              const float* __restrict__ Cin,
                              const float* __restrict__ bias,
                              float* __restrict__ D,
                              int M, int N, int K, int relu, int permRows)
{
    const int lane   = threadIdx.x & 31;
    const int wave   = threadIdx.x >> 5;
    const int strips = N >> 6;                 // 64-wide N strips
    const int tilesM = M >> 4;
    int tile = blockIdx.x * (blockDim.x >> 5) + wave;
    if (tile >= tilesM * strips) return;       // uniform per-wave, EXEC stays full

    const int tm = tile / strips;
    const int ts = tile - tm * strips;
    const int m0 = tm << 4, n0 = ts << 6;
    const int g  = lane >> 4;                  // half-wave group
    const int ln = lane & 15;

    // --- C/D fragments (ISA 7.12.2: VGPR r -> M = m0 + r + 8*g, N = n' + ln) ---
    v8f acc[4];
#pragma unroll
    for (int i = 0; i < 4; ++i) {
        if (Cin) {
            const float* cp = Cin + (size_t)(m0 + (g << 3)) * N + (n0 + (i << 4) + ln);
#pragma unroll
            for (int r = 0; r < 8; ++r) acc[i][r] = cp[(size_t)r * N];
        } else {
#pragma unroll
            for (int r = 0; r < 8; ++r) acc[i][r] = 0.0f;
        }
    }

    const _Float16* arow = A + (size_t)(m0 + ln) * K;   // A row m = m0 + ln

    for (int k0 = 0; k0 < K; k0 += 32) {
        // A 16x32 f16 frag: lanes0-15 K {0..15}, lanes16-31 K {16..31}
        // => two contiguous 8-half (16B) chunks per lane.
        v8h alo = *(const v8h*)(arow + k0 + (g << 4));
        v8h ahi = *(const v8h*)(arow + k0 + 8 + (g << 4));
        v16h a;
#pragma unroll
        for (int j = 0; j < 8; ++j) { a[j] = alo[j]; a[8 + j] = ahi[j]; }

        if (k0 + 32 < K) __builtin_prefetch(arow + k0 + 32, 0, 1);

#pragma unroll
        for (int i = 0; i < 4; ++i) {
            // B 32x16 frag: lane ln = column n, halves K = k0+g*16 .. +15 contiguous (32B)
            const v16h b = *(const v16h*)(W + (size_t)(n0 + (i << 4) + ln) * K + (g << 4) + k0);
            acc[i] = __builtin_amdgcn_wmma_f32_16x16x32_f16(false, a, false, b,
                                                            (short)0, acc[i], false, false);
        }
    }

#pragma unroll
    for (int i = 0; i < 4; ++i) {
        const int n = n0 + (i << 4) + ln;
        const float bv = bias ? bias[n] : 0.0f;
#pragma unroll
        for (int r = 0; r < 8; ++r) {
            int row = m0 + (g << 3) + r;
            float v = acc[i][r] + bv;
            if (relu) v = fmaxf(v, 0.0f);
            size_t offo;
            if (permRows) {
                int bb = row & 31, tt = row >> 5;             // row = t*32 + b
                offo = ((size_t)(bb * T_DECL + tt)) * N + n;  // -> out[b, t, n]
            } else {
                offo = (size_t)row * N + n;
            }
            D[offo] = v;
        }
    }
}

// one-time f32 -> f16 conversion (weights)
__global__ void cvt_f16_kernel(const float* __restrict__ src,
                               _Float16* __restrict__ dst, long long n)
{
    long long gid    = (long long)blockIdx.x * blockDim.x + threadIdx.x;
    long long stride = (long long)gridDim.x * blockDim.x;
    for (; gid < n; gid += stride) dst[gid] = (_Float16)src[gid];
}

// ---------------------------------------------------------------------------
// Encoder embedding gather -> f16 [T,B,E] (forward + per-sequence reversed)
// ---------------------------------------------------------------------------
__global__ void gather_emb_kernel(const int* __restrict__ enc_in,
                                  const int* __restrict__ enc_len,
                                  const float* __restrict__ emb_f,
                                  _Float16* __restrict__ x_f, _Float16* __restrict__ x_r)
{
    int gid = blockIdx.x * blockDim.x + threadIdx.x;
    if (gid >= B_SZ * T_ENCL * EMB) return;
    int j = gid & 511;
    int t = (gid >> 9) & 63;
    int b = gid >> 15;
    int len = enc_len[b];
    int rid = (t < len) ? (len - 1 - t) : t;
    x_f[(size_t)(t * B_SZ + b) * EMB + j] =
        (_Float16)emb_f[(size_t)enc_in[b * T_ENCL + t] * EMB + j];
    x_r[(size_t)(t * B_SZ + b) * EMB + j] =
        (_Float16)emb_f[(size_t)enc_in[b * T_ENCL + rid] * EMB + j];
}

// zero recurrent state: h (f16, both dirs) and c (f32, both dirs)
__global__ void init_state_kernel(_Float16* __restrict__ h16, float* __restrict__ c)
{
    int gid = blockIdx.x * blockDim.x + threadIdx.x;
    if (gid >= 2 * B_SZ * ENCH) return;
    h16[gid] = (_Float16)0.0f;
    c[gid]   = 0.0f;
}

__global__ void bias_sum_kernel(const float* bi_f, const float* bh_f,
                                const float* bi_b, const float* bh_b,
                                float* bsf, float* bsb)
{
    int gid = blockIdx.x * blockDim.x + threadIdx.x;
    if (gid < 2048)      bsf[gid] = bi_f[gid] + bh_f[gid];
    else if (gid < 4096) { int j = gid - 2048; bsb[j] = bi_b[j] + bh_b[j]; }
}

// ---------------------------------------------------------------------------
// LSTM pointwise: gates -> (h16,c) update + masked packed-seq output (both dirs)
// ---------------------------------------------------------------------------
__global__ void lstm_pointwise(const float* __restrict__ gates_f,
                               const float* __restrict__ gates_b,
                               _Float16* __restrict__ h_f16, float* __restrict__ c_f,
                               _Float16* __restrict__ h_b16, float* __restrict__ c_b,
                               float* __restrict__ hall_f, float* __restrict__ hall_b,
                               const int* __restrict__ enc_len, int t)
{
    int gid = blockIdx.x * blockDim.x + threadIdx.x;
    if (gid >= 2 * B_SZ * ENCH) return;
    int dir = gid >> 14;
    int rem = gid & 16383;
    int b = rem >> 9;
    int j = rem & 511;
    const float* gates = dir ? gates_b : gates_f;
    _Float16* h = dir ? h_b16 : h_f16;
    float* c    = dir ? c_b : c_f;
    float* hall = dir ? hall_b : hall_f;

    float gi = gates[b * 2048 + j];           // i,f,g,o gate order
    float gf = gates[b * 2048 + 512 + j];
    float gg = gates[b * 2048 + 1024 + j];
    float go = gates[b * 2048 + 1536 + j];
    float c_old = c[b * 512 + j];
    float c_new = sigm(gf) * c_old + sigm(gi) * tanhf(gg);
    float h_new = sigm(go) * tanhf(c_new);
    bool m = t < enc_len[b];
    hall[(size_t)(t * B_SZ + b) * 512 + j] = m ? h_new : 0.0f;
    if (m) {                                  // packed-seq: state frozen past length
        h[b * 512 + j] = (_Float16)h_new;
        c[b * 512 + j] = c_new;
    }
}

// enc_out[b,t] = [ h_fwd[t,b] ; h_bwd_rev[rev_idx(b,t), b] ]
__global__ void assemble_enc_out(const float* __restrict__ hall_f,
                                 const float* __restrict__ hall_b,
                                 const int* __restrict__ enc_len,
                                 float* __restrict__ enc_out)
{
    int gid = blockIdx.x * blockDim.x + threadIdx.x;
    if (gid >= B_SZ * T_ENCL * ENCH) return;
    int j = gid & 511;
    int t = (gid >> 9) & 63;
    int b = gid >> 15;
    int len = enc_len[b];
    int rid = (t < len) ? (len - 1 - t) : t;
    enc_out[(size_t)(b * T_ENCL + t) * 1024 + j]       = hall_f[(size_t)(t * B_SZ + b) * 512 + j];
    enc_out[(size_t)(b * T_ENCL + t) * 1024 + 512 + j] = hall_b[(size_t)(rid * B_SZ + b) * 512 + j];
}

// enc_feat[b,t,:8] = enc_out[b,t] @ Wa[0:1024,:] + ba   (precomputed once)
__global__ void enc_feat_kernel(const float* __restrict__ enc_out,
                                const float* __restrict__ Wa,
                                const float* __restrict__ ba,
                                float* __restrict__ enc_feat)
{
    int gid = blockIdx.x * blockDim.x + threadIdx.x;
    if (gid >= B_SZ * T_ENCL * 8) return;
    int j = gid & 7;
    int t = (gid >> 3) & 63;
    int b = gid >> 9;
    const float* eo = enc_out + (size_t)(b * T_ENCL + t) * 1024;
    float s = ba[j];
    for (int k = 0; k < 1024; ++k) s += eo[k] * Wa[(size_t)k * 8 + j];
    enc_feat[(size_t)(b * T_ENCL + t) * 8 + j] = s;
}

// ---------------------------------------------------------------------------
// Per decoder step: concat-attention (masked softmax) + xin16 = [emb_e ; ctx]
// One block per batch element.
// ---------------------------------------------------------------------------
__global__ void attn_xin_kernel(const float* __restrict__ enc_out,
                                const float* __restrict__ enc_feat,
                                const float* __restrict__ Wa,
                                const float* __restrict__ va,
                                const float* __restrict__ state,   // null at t==0
                                const int* __restrict__ enc_len,
                                const int* __restrict__ dec_in,
                                const float* __restrict__ emb_e,
                                _Float16* __restrict__ xin16, int t)
{
    __shared__ float dp[8];
    __shared__ float sc[T_ENCL];
    __shared__ float red[256];
    const int b = blockIdx.x;
    const int tid = threadIdx.x;

    if (state) {                                  // dproj = state[b] @ Wa_dec
        int j  = tid & 7;
        int sl = tid >> 3;                        // 32 k-slices of 32
        const float* st = state + (size_t)b * DECH;
        float s = 0.0f;
        for (int k = sl * 32; k < sl * 32 + 32; ++k)
            s += st[k] * Wa[(size_t)(2 * ENCH + k) * 8 + j];
        red[tid] = s;
        __syncthreads();
        if (tid < 8) {
            float s2 = 0.0f;
            for (int i = 0; i < 32; ++i) s2 += red[tid + (i << 3)];
            dp[tid] = s2;
        }
    } else {
        if (tid < 8) dp[tid] = 0.0f;
    }
    __syncthreads();

    if (tid < T_ENCL) {                           // masked scores
        const float* f = enc_feat + (size_t)(b * T_ENCL + tid) * 8;
        float s = 0.0f;
#pragma unroll
        for (int j = 0; j < 8; ++j) s += va[j] * tanhf(f[j] + dp[j]);
        sc[tid] = (tid < enc_len[b]) ? s : -1e9f;
    }
    __syncthreads();

    if (tid == 0) {                               // softmax over 64 (tiny)
        float m = sc[0];
        for (int i = 1; i < T_ENCL; ++i) m = fmaxf(m, sc[i]);
        float sum = 0.0f;
        for (int i = 0; i < T_ENCL; ++i) { float e = __expf(sc[i] - m); sc[i] = e; sum += e; }
        float inv = 1.0f / sum;
        for (int i = 0; i < T_ENCL; ++i) sc[i] *= inv;
    }
    __syncthreads();

    for (int d = tid; d < 2 * ENCH; d += blockDim.x) {   // context
        float cx = 0.0f;
        const float* eo = enc_out + (size_t)(b * T_ENCL) * 1024 + d;
        for (int tt = 0; tt < T_ENCL; ++tt) cx += sc[tt] * eo[(size_t)tt * 1024];
        xin16[(size_t)b * 1536 + 512 + d] = (_Float16)cx;
    }
    int tok = dec_in[b * T_DECL + t];
    for (int e = tid; e < 512; e += blockDim.x)
        xin16[(size_t)b * 1536 + e] = (_Float16)emb_e[(size_t)tok * 512 + e];
}

// GRU with hx==0: h = (1-z)*n ; state stored f32 (attention) + f16 (classifier A)
__global__ void gru_pointwise(const float* __restrict__ gi,
                              const float* __restrict__ bh,
                              float* __restrict__ H_all,
                              _Float16* __restrict__ Hall16, int t)
{
    int gid = blockIdx.x * blockDim.x + threadIdx.x;
    if (gid >= B_SZ * DECH) return;
    int j = gid & 1023;
    int b = gid >> 10;
    float r = sigm(gi[(size_t)b * 3072 + j]        + bh[j]);
    float z = sigm(gi[(size_t)b * 3072 + 1024 + j] + bh[1024 + j]);
    float n = tanhf(gi[(size_t)b * 3072 + 2048 + j] + r * bh[2048 + j]);
    float h = (1.0f - z) * n;
    H_all [(size_t)(t * B_SZ + b) * DECH + j] = h;
    Hall16[(size_t)(t * B_SZ + b) * DECH + j] = (_Float16)h;
}

// ---------------------------------------------------------------------------
extern "C" void kernel_launch(void* const* d_in, const int* in_sizes, int n_in,
                              void* d_out, int out_size, void* d_ws, size_t ws_size,
                              hipStream_t stream)
{
    const int*   enc_in  = (const int*)  d_in[0];
    const int*   enc_len = (const int*)  d_in[1];
    const int*   dec_in  = (const int*)  d_in[2];
    /* d_in[3] decoder_seq_length: unused (decoder always runs T_DEC steps) */
    const float* emb_f   = (const float*)d_in[4];
    const float* emb_e   = (const float*)d_in[5];
    const float* W_ih_f  = (const float*)d_in[6];
    const float* W_hh_f  = (const float*)d_in[7];
    const float* b_ih_f  = (const float*)d_in[8];
    const float* b_hh_f  = (const float*)d_in[9];
    const float* W_ih_b  = (const float*)d_in[10];
    const float* W_hh_b  = (const float*)d_in[11];
    const float* b_ih_b  = (const float*)d_in[12];
    const float* b_hh_b  = (const float*)d_in[13];
    const float* Wa      = (const float*)d_in[14];
    const float* ba      = (const float*)d_in[15];
    const float* va      = (const float*)d_in[16];
    const float* Wi_gru  = (const float*)d_in[17];
    /* d_in[18] Wh_gru: dead (GRU hidden is zero) */
    const float* bi_gru  = (const float*)d_in[19];
    const float* bh_gru  = (const float*)d_in[20];
    const float* Wc      = (const float*)d_in[21];
    const float* bc      = (const float*)d_in[22];
    float* out = (float*)d_out;

    char* ws = (char*)d_ws;
    size_t off = 0;
    auto alloc_f = [&](size_t n) -> float* {
        off = (off + 63) & ~(size_t)63;
        float* p = (float*)(ws + off); off += n * sizeof(float); return p;
    };
    auto alloc_h = [&](size_t n) -> _Float16* {
        off = (off + 63) & ~(size_t)63;
        _Float16* p = (_Float16*)(ws + off); off += n * sizeof(_Float16); return p;
    };

    // f16 operands for WMMA GEMMs
    _Float16* x_f16   = alloc_h((size_t)T_ENCL * B_SZ * EMB);
    _Float16* x_r16   = alloc_h((size_t)T_ENCL * B_SZ * EMB);
    _Float16* Wihf16  = alloc_h((size_t)2048 * EMB);
    _Float16* Wihb16  = alloc_h((size_t)2048 * EMB);
    _Float16* Whhf16  = alloc_h((size_t)2048 * ENCH);
    _Float16* Whhb16  = alloc_h((size_t)2048 * ENCH);
    _Float16* Wig16   = alloc_h((size_t)3072 * 1536);
    _Float16* Wc16    = alloc_h((size_t)VOUT * DECH);
    _Float16* h_f16   = alloc_h((size_t)B_SZ * ENCH);      // h state (both dirs,
    _Float16* h_b16   = alloc_h((size_t)B_SZ * ENCH);      //  contiguous)
    _Float16* xin16   = alloc_h((size_t)B_SZ * 1536);
    _Float16* Hall16  = alloc_h((size_t)T_DECL * B_SZ * DECH);
    // f32 buffers
    float* XWf     = alloc_f((size_t)T_ENCL * B_SZ * 2048);   // x @ W_ih^T + biases
    float* XWb     = alloc_f((size_t)T_ENCL * B_SZ * 2048);
    float* gates_f = alloc_f((size_t)B_SZ * 2048);
    float* gates_b = alloc_f((size_t)B_SZ * 2048);
    float* c_f     = alloc_f((size_t)B_SZ * ENCH);            // c state (both dirs,
    float* c_b     = alloc_f((size_t)B_SZ * ENCH);            //  contiguous)
    float* hall_f  = alloc_f((size_t)T_ENCL * B_SZ * ENCH);
    float* hall_b  = alloc_f((size_t)T_ENCL * B_SZ * ENCH);
    float* enc_out = alloc_f((size_t)B_SZ * T_ENCL * 1024);
    float* encfeat = alloc_f((size_t)B_SZ * T_ENCL * 8);
    float* bsf     = alloc_f(2048);
    float* bsb     = alloc_f(2048);
    float* gi      = alloc_f((size_t)B_SZ * 3072);
    float* H_all   = alloc_f((size_t)T_DECL * B_SZ * DECH);
    (void)ws_size; (void)in_sizes; (void)n_in; (void)out_size;

    auto gemm = [&](const _Float16* A, const _Float16* W, const float* Cin,
                    const float* bias, float* D, int M, int N, int K, int relu, int perm) {
        int tiles  = (M / 16) * (N / 64);                    // 16x64 per wave
        int blocks = (tiles + 7) / 8;                        // 8 waves / block
        wmma_gemm_f16<<<blocks, 256, 0, stream>>>(A, W, Cin, bias, D, M, N, K, relu, perm);
    };
    auto cvt = [&](const float* s, _Float16* d, long long n) {
        cvt_f16_kernel<<<2048, 256, 0, stream>>>(s, d, n);
    };

    // --- one-time weight conversion to f16 ---
    cvt(W_ih_f, Wihf16, 2048LL * EMB);
    cvt(W_ih_b, Wihb16, 2048LL * EMB);
    cvt(W_hh_f, Whhf16, 2048LL * ENCH);
    cvt(W_hh_b, Whhb16, 2048LL * ENCH);
    cvt(Wi_gru, Wig16,  3072LL * 1536);
    cvt(Wc,     Wc16,   (long long)VOUT * DECH);

    // --- encoder setup ---
    bias_sum_kernel<<<16, 256, 0, stream>>>(b_ih_f, b_hh_f, b_ih_b, b_hh_b, bsf, bsb);
    gather_emb_kernel<<<4096, 256, 0, stream>>>(enc_in, enc_len, emb_f, x_f16, x_r16);
    init_state_kernel<<<128, 256, 0, stream>>>(h_f16, c_f);   // h_f16+h_b16, c_f+c_b

    // batched input-side LSTM GEMMs (biases folded here)
    gemm(x_f16, Wihf16, nullptr, bsf, XWf, T_ENCL * B_SZ, 2048, EMB, 0, 0);
    gemm(x_r16, Wihb16, nullptr, bsb, XWb, T_ENCL * B_SZ, 2048, EMB, 0, 0);

    // --- recurrent LSTM loop: only h @ W_hh^T stays sequential ---
    for (int t = 0; t < T_ENCL; ++t) {
        gemm(h_f16, Whhf16, XWf + (size_t)t * B_SZ * 2048, nullptr, gates_f, B_SZ, 2048, ENCH, 0, 0);
        gemm(h_b16, Whhb16, XWb + (size_t)t * B_SZ * 2048, nullptr, gates_b, B_SZ, 2048, ENCH, 0, 0);
        lstm_pointwise<<<128, 256, 0, stream>>>(gates_f, gates_b, h_f16, c_f, h_b16, c_b,
                                                hall_f, hall_b, enc_len, t);
    }

    assemble_enc_out<<<4096, 256, 0, stream>>>(hall_f, hall_b, enc_len, enc_out);
    enc_feat_kernel<<<64, 256, 0, stream>>>(enc_out, Wa, ba, encfeat);

    // --- decoder loop: attention + GRU-input GEMM; logits deferred ---
    for (int t = 0; t < T_DECL; ++t) {
        const float* state = (t == 0) ? nullptr : (H_all + (size_t)(t - 1) * B_SZ * DECH);
        attn_xin_kernel<<<B_SZ, 256, 0, stream>>>(enc_out, encfeat, Wa, va, state,
                                                  enc_len, dec_in, emb_e, xin16, t);
        gemm(xin16, Wig16, nullptr, bi_gru, gi, B_SZ, 3072, 1536, 0, 0);
        gru_pointwise<<<128, 256, 0, stream>>>(gi, bh_gru, H_all, Hall16, t);
    }

    // --- one big WMMA classifier GEMM over all 1024 decoder states ---
    // [1024,1024](f16) x [1024,32000](f16) + bc, ReLU, rows scattered to out[b,t,:]
    gemm(Hall16, Wc16, nullptr, bc, out, T_DECL * B_SZ, VOUT, DECH, 1, 1);
}
